// HybridModel_56959856280435
// MI455X (gfx1250) — matrix operand order repaired and checked by reading, not compiled, not run on hardware
//
#include <hip/hip_runtime.h>

typedef __attribute__((ext_vector_type(16))) _Float16 v16h;
typedef __attribute__((ext_vector_type(8)))  float    v8f;

constexpr int kBatch   = 4096;
constexpr int kD       = 64;    // species
constexpr int kW       = 128;   // MLP width
constexpr int kR       = 128;   // reactions
constexpr int kT       = 128;   // save points
constexpr int kNSub    = 4;     // Tsit5 substeps per interval
constexpr int kMT      = 16;    // batch rows per workgroup (one M tile)
constexpr int kThreads = 128;   // 4 waves (wave32)

// padded LDS strides (in halfs) for conflict-free A-fragment gathers
constexpr int kYStr  = kD  + 4;   // 68
constexpr int kH1Str = kW  + 4;   // 132
constexpr int kHFStr = 2*kW + 4;  // 260

// ---- Tsit5 tableau (constexpr so unrolled stages fold to literal constants) ----
constexpr float AC[15] = {
  0.161f,
  -0.008480655492356989f, 0.335480655492357f,
  2.8971530571054935f, -6.359448489975075f, 4.3622954328695815f,
  5.325864828439257f, -11.748883564062828f, 7.4955393428898365f, -0.09249506636175525f,
  5.86145544294642f, -12.92096931784711f, 8.159367898576159f, -0.071584973281401f, -0.028269050394068383f
};
constexpr int ACOFF[6] = {0, 0, 1, 3, 6, 10};
constexpr float BC[6] = {
  0.09646076681806523f, 0.01f, 0.4798896504144996f,
  1.379008574103742f, -3.290069515436081f, 2.324710524099774f
};

static __device__ __forceinline__ v8f wmma16(v16h a, v16h b, v8f c) {
  return __builtin_amdgcn_wmma_f32_16x16x32_f16(false, a, false, b, (short)0, c, false, false);
}

// Hardware tanh on gfx1250 (TRANS32); fallback is 2 TRANS ops (exp2 + rcp).
static __device__ __forceinline__ float fast_tanh(float x) {
#if __has_builtin(__builtin_amdgcn_tanhf)
  return __builtin_amdgcn_tanhf(x);
#else
  float t = __builtin_amdgcn_exp2f(__builtin_fabsf(x) * -2.885390081777927f); // exp(-2|x|)
  float r = (1.0f - t) * __builtin_amdgcn_rcpf(1.0f + t);
  return __builtin_copysignf(r, x);
#endif
}

// A-fragment (16x32 f16) gather from row-major f16 LDS buffer with padded stride.
// lane m=l&15, K base=(l>>4)*16; VGPR v holds K = base + v*2 {+0,+1} (matches B contiguous order)
static __device__ __forceinline__ v16h load_A(const _Float16* buf, int stride, int kbase, int lane) {
  const int m  = lane & 15;
  const int kb = kbase + ((lane >> 4) << 4);
  union { v16h h; unsigned u[8]; } r;
  const _Float16* row = buf + m * stride;
#pragma unroll
  for (int v = 0; v < 8; ++v) {
    int k = kb + (v << 1);
    r.u[v] = *(const unsigned*)(row + k);  // 4B aligned: stride & k even
  }
  return r.h;
}

// Gather one WMMA B-fragment (32x16 f16) for this lane directly from global fp32
// weights (one-time, L2-resident). B layout: lane -> column n = l&15, K half =
// (l>>4)*16, element j -> K = khalf + j.
template <typename F>
static __device__ __forceinline__ v16h gather_Bfrag(F src, int kt, int nt, int lane) {
  const int n  = nt * 16 + (lane & 15);
  const int kb = kt * 32 + ((lane >> 4) << 4);
  v16h r;
#pragma unroll
  for (int j = 0; j < 16; ++j)
    r[j] = (_Float16)src(kb + j, n);
  return r;
}

__global__ __launch_bounds__(kThreads)
void node_tsit5_kernel(const float* __restrict__ ts, const float* __restrict__ y0,
                       const float* __restrict__ W1g, const float* __restrict__ b1g,
                       const float* __restrict__ W2g, const float* __restrict__ b2g,
                       const float* __restrict__ W3g, const float* __restrict__ b3g,
                       const float* __restrict__ kg,  const float* __restrict__ Sg,
                       const int*   __restrict__ subg, float* __restrict__ out) {
  __shared__ float sB1[kW], sB2[kW], sB3[kD], sKc[kR];
  __shared__ int   sSub[kR];
  __shared__ float sTs[kT];
  __shared__ float sY[kMT * kD];      // current state (fp32)
  __shared__ float sYt[kMT * kD];     // stage input y-temp (fp32, for flux gather)
  __shared__ __attribute__((aligned(16))) _Float16 sYh[kMT * kYStr];  // y-temp f16 (GEMM1 A)
  __shared__ __attribute__((aligned(16))) _Float16 sH1[kMT * kH1Str]; // hidden1 f16 (GEMM2 A)
  __shared__ __attribute__((aligned(16))) _Float16 sHF[kMT * kHFStr]; // [H2 | flux] f16 (GEMM3 A)
  __shared__ float sKst[6][kMT * kD]; // Tsit5 stage derivatives

  const int tid  = threadIdx.x;
  const int lane = tid & 31;
  const int wave = tid >> 5;
  const int row0 = blockIdx.x * kMT;

  // ---- weight B-fragments resident in VGPRs for the whole integration ----
  // per wave: 4 (W1) + 8 (W2) + 8 (W3|S) fragments = 160 VGPRs
  v16h w1f[2][2], w2f[2][4], w34f[8];
#pragma unroll
  for (int i2 = 0; i2 < 2; ++i2) {
    const int nt = 2 * wave + i2;
#pragma unroll
    for (int kt = 0; kt < 2; ++kt)
      w1f[i2][kt] = gather_Bfrag([&](int k, int n) { return W1g[n * kD + k]; }, kt, nt, lane);
#pragma unroll
    for (int kt = 0; kt < 4; ++kt)
      w2f[i2][kt] = gather_Bfrag([&](int k, int n) { return W2g[n * kW + k]; }, kt, nt, lane);
  }
#pragma unroll
  for (int kt = 0; kt < 8; ++kt)
    w34f[kt] = gather_Bfrag([&](int k, int n) {
      return (k < kW) ? W3g[n * kW + k] : Sg[n * kR + (k - kW)];
    }, kt, wave, lane);

  // ---- scalars / state into LDS ----
  for (int i = tid; i < kW; i += kThreads) {
    sB1[i] = b1g[i]; sB2[i] = b2g[i]; sKc[i] = kg[i]; sSub[i] = subg[i];
  }
  for (int i = tid; i < kD; i += kThreads) sB3[i] = b3g[i];
  for (int i = tid; i < kT; i += kThreads) sTs[i] = ts[i];
  for (int i = tid; i < kMT * kD; i += kThreads) {
    float v = y0[(size_t)row0 * kD + i];
    sY[i] = v;
    __builtin_nontemporal_store(v, out + (size_t)row0 * kD + i);  // t = 0 snapshot
  }
  __syncthreads();

  for (int t = 0; t < kT - 1; ++t) {
    const float dt = (sTs[t + 1] - sTs[t]) * (1.0f / kNSub);
    for (int sub = 0; sub < kNSub; ++sub) {
#pragma unroll
      for (int s = 0; s < 6; ++s) {
        // ---- stage input: ytmp = y + dt * sum(a_sj * k_j)  (literal coeffs) ----
        for (int i = tid; i < kMT * kD; i += kThreads) {
          float acc = sY[i];
#pragma unroll
          for (int j = 0; j < s; ++j) acc += dt * AC[ACOFF[s] + j] * sKst[j][i];
          sYt[i] = acc;
          sYh[(i >> 6) * kYStr + (i & 63)] = (_Float16)acc;
        }
        __syncthreads();

        // ---- GEMM1: H1 = tanh(Ytmp[16x64] @ W1t[64x128] + b1) ----
        {
          v16h a0 = load_A(sYh, kYStr, 0, lane);
          v16h a1 = load_A(sYh, kYStr, 32, lane);
#pragma unroll
          for (int i2 = 0; i2 < 2; ++i2) {
            v8f c = {};
            c = wmma16(a0, w1f[i2][0], c);
            c = wmma16(a1, w1f[i2][1], c);
            const int n = (2 * wave + i2) * 16 + (lane & 15);
            const float bias = sB1[n];
            const int mb = (lane >> 4) << 3;
#pragma unroll
            for (int v = 0; v < 8; ++v)
              sH1[(mb + v) * kH1Str + n] = (_Float16)fast_tanh(c[v] + bias);
          }
        }
        // ---- flux_j = k_j * ytmp[:, sub_idx_j] -> HF[:,128:256]  (balances phase) ----
        for (int i = tid; i < kMT * kR; i += kThreads) {
          const int m = i >> 7, r = i & (kR - 1);
          sHF[m * kHFStr + kW + r] = (_Float16)(sKc[r] * sYt[(m << 6) + sSub[r]]);
        }
        __syncthreads();

        // ---- GEMM2: H2 = tanh(H1[16x128] @ W2t[128x128] + b2) -> HF[:,0:128] ----
        {
          v16h a[4];
#pragma unroll
          for (int kt = 0; kt < 4; ++kt) a[kt] = load_A(sH1, kH1Str, kt * 32, lane);
#pragma unroll
          for (int i2 = 0; i2 < 2; ++i2) {
            v8f c = {};
#pragma unroll
            for (int kt = 0; kt < 4; ++kt)
              c = wmma16(a[kt], w2f[i2][kt], c);
            const int n = (2 * wave + i2) * 16 + (lane & 15);
            const float bias = sB2[n];
            const int mb = (lane >> 4) << 3;
#pragma unroll
            for (int v = 0; v < 8; ++v)
              sHF[(mb + v) * kHFStr + n] = (_Float16)fast_tanh(c[v] + bias);
          }
        }
        __syncthreads();

        // ---- GEMM3: k_s = [H2|flux][16x256] @ [[W3t];[St]][256x64] + b3 ----
        {
          v8f c = {};
#pragma unroll
          for (int kt = 0; kt < 8; ++kt) {
            v16h a = load_A(sHF, kHFStr, kt * 32, lane);
            c = wmma16(a, w34f[kt], c);
          }
          const int n = wave * 16 + (lane & 15);
          const float bias = sB3[n];
          const int mb = (lane >> 4) << 3;
#pragma unroll
          for (int v = 0; v < 8; ++v)
            sKst[s][(mb + v) * kD + n] = c[v] + bias;
        }
        __syncthreads();
      }
      // ---- y += dt * sum(b_j * k_j) ----
      for (int i = tid; i < kMT * kD; i += kThreads) {
        float acc = sY[i];
#pragma unroll
        for (int j = 0; j < 6; ++j) acc += dt * BC[j] * sKst[j][i];
        sY[i] = acc;
      }
      __syncthreads();
    }
    // coalesced, non-temporal 4KB snapshot write per block (write-once data)
    for (int i = tid; i < kMT * kD; i += kThreads)
      __builtin_nontemporal_store(sY[i], out + ((size_t)(t + 1) * kBatch + row0) * kD + i);
  }
}

extern "C" void kernel_launch(void* const* d_in, const int* in_sizes, int n_in,
                              void* d_out, int out_size, void* d_ws, size_t ws_size,
                              hipStream_t stream) {
  (void)in_sizes; (void)n_in; (void)out_size; (void)d_ws; (void)ws_size;
  const float* ts  = (const float*)d_in[0];
  const float* y0  = (const float*)d_in[1];
  const float* W1  = (const float*)d_in[2];
  const float* b1  = (const float*)d_in[3];
  const float* W2  = (const float*)d_in[4];
  const float* b2  = (const float*)d_in[5];
  const float* W3  = (const float*)d_in[6];
  const float* b3  = (const float*)d_in[7];
  const float* k   = (const float*)d_in[8];
  const float* S   = (const float*)d_in[9];
  const int*   sub = (const int*)d_in[10];
  float* out = (float*)d_out;

  dim3 grid(kBatch / kMT);   // 256 workgroups, each owns 16 batch rows
  dim3 block(kThreads);      // 4 waves of 32
  hipLaunchKernelGGL(node_tsit5_kernel, grid, block, 0, stream,
                     ts, y0, W1, b1, W2, b2, W3, b3, k, S, sub, out);
}
